// GINModel_70334384439968
// MI455X (gfx1250) — matrix operand order repaired and checked by reading, not compiled, not run on hardware
//
#include <hip/hip_runtime.h>

// ---------------------------------------------------------------------------
// GIN forward for MI455X (gfx1250, wave32):
//   h1 = MLP1(x + scatter_sum(x));  h2 = MLP2(h1 + scatter_sum(h1))
//   out = mean(h2) @ Wfc + bfc
// GEMMs use v_wmma_f32_16x16x32_bf16 (bf16 A/B, f32 accum).
// Aggregation uses explicit inline-asm global_atomic_add_f32 (scope DEV) so
// the 400M scatter-adds hit the L2 atomic units (never a CAS loop).
// ---------------------------------------------------------------------------

typedef __attribute__((ext_vector_type(16))) __bf16 v16bf;
typedef __attribute__((ext_vector_type(8)))  float  v8f;

#define DFEAT 128

// Hardware f32 atomic add, non-returning, device scope (L2 atomic units).
__device__ __forceinline__ void atomic_add_f32_dev(float* p, float v) {
    asm volatile("global_atomic_add_f32 %0, %1, off scope:SCOPE_DEV"
                 :: "v"(p), "v"(v)
                 : "memory");
}

// ---- weight prep: wt[n*128+k] = (bf16) W[k*128+n]  (transpose + convert) ----
__global__ void k_prep_wt(const float* __restrict__ W, __bf16* __restrict__ wt) {
    int i = blockIdx.x * blockDim.x + threadIdx.x;   // 0..16383
    if (i >= DFEAT * DFEAT) return;
    int n = i >> 7;          // output column of W
    int k = i & 127;         // input (K) index
    wt[i] = (__bf16)W[k * DFEAT + n];
}

// ---- copy h -> out (self term of GIN, eps=0) ----
__global__ void k_copy(const float4* __restrict__ src, float4* __restrict__ dst, int n4) {
    int i = blockIdx.x * blockDim.x + threadIdx.x;
    if (i < n4) dst[i] = src[i];
}

// ---- scatter: out[dst] += h[src], one edge per wave, 4 floats per lane ----
__global__ void k_scatter(const float* __restrict__ h,
                          const long long* __restrict__ src,
                          const long long* __restrict__ dst,
                          float* __restrict__ out, int E) {
    int t    = blockIdx.x * blockDim.x + threadIdx.x;
    int e    = t >> 5;                 // wave-uniform edge index
    int lane = t & 31;
    if (e >= E) return;
    long long s = src[e];
    long long d = dst[e];
    float4 v = *(const float4*)(h + (size_t)s * DFEAT + lane * 4);
    float* o = out + (size_t)d * DFEAT + lane * 4;
    atomic_add_f32_dev(o + 0, v.x);
    atomic_add_f32_dev(o + 1, v.y);
    atomic_add_f32_dev(o + 2, v.z);
    atomic_add_f32_dev(o + 3, v.w);
}

// ---- WMMA GEMM: out[r,c] = act( in[r,:] @ W + bias[c] ), W given as wt[n][k] bf16 ----
// Block: 256 threads = 8 waves; each wave computes a 16(M) x 128(N) strip.
template <bool RELU>
__global__ __launch_bounds__(256) void k_gemm(const float* __restrict__ in,
                                              const __bf16* __restrict__ wt,
                                              const float* __restrict__ bias,
                                              float* __restrict__ out, int nrows) {
    __shared__ __bf16 sW[DFEAT * DFEAT];             // 32 KB of the 320 KB WGP LDS

    // Cooperative stage of the whole 128x128 bf16 weight tile into LDS.
    {
        const uint4* g = (const uint4*)wt;
        uint4*       s = (uint4*)sW;
        for (int i = threadIdx.x; i < (DFEAT * DFEAT * 2) / 16; i += 256) s[i] = g[i];
    }
    __syncthreads();

    const int wave = threadIdx.x >> 5;
    const int lane = threadIdx.x & 31;
    const int tile = blockIdx.x * 8 + wave;          // 16-row tile index
    if (tile * 16 >= nrows) return;                  // whole-wave early exit (post-barrier)

    const int rowbase = tile * 16;
    const int half    = lane >> 4;                   // 0: lanes 0-15, 1: lanes 16-31
    const int lm      = lane & 15;

    const float* arow = in + (size_t)(rowbase + lm) * DFEAT;

    // 8 accumulator tiles (16x16 each) spanning the 128 output columns,
    // pre-seeded with the bias (bias depends only on the column = lm lane pos).
    v8f acc[8];
#pragma unroll
    for (int nt = 0; nt < 8; ++nt) {
        float bv = bias[nt * 16 + lm];
        acc[nt] = (v8f){bv, bv, bv, bv, bv, bv, bv, bv};
    }

#pragma unroll
    for (int k0 = 0; k0 < DFEAT; k0 += 32) {
        // A fragment, 16x32 bf16 (contiguous K, matching B):
        //   lane half 0 -> K = k0+0..15
        //   lane half 1 -> K = k0+16..31
        const int kA = k0 + half * 16;
        float4 f0 = *(const float4*)(arow + kA);
        float4 f1 = *(const float4*)(arow + kA + 4);
        float4 f2 = *(const float4*)(arow + kA + 8);
        float4 f3 = *(const float4*)(arow + kA + 12);
        v16bf a;
        a[0]  = (__bf16)f0.x; a[1]  = (__bf16)f0.y; a[2]  = (__bf16)f0.z; a[3]  = (__bf16)f0.w;
        a[4]  = (__bf16)f1.x; a[5]  = (__bf16)f1.y; a[6]  = (__bf16)f1.z; a[7]  = (__bf16)f1.w;
        a[8]  = (__bf16)f2.x; a[9]  = (__bf16)f2.y; a[10] = (__bf16)f2.z; a[11] = (__bf16)f2.w;
        a[12] = (__bf16)f3.x; a[13] = (__bf16)f3.y; a[14] = (__bf16)f3.z; a[15] = (__bf16)f3.w;

#pragma unroll
        for (int nt = 0; nt < 8; ++nt) {
            // B fragment, 32x16 bf16: lane holds 16 contiguous K values of one column.
            //   lanes 0-15: col=lm, K=k0..k0+15 ; lanes 16-31: col=lm, K=k0+16..k0+31
            const int col = nt * 16 + lm;
            v16bf b = *(const v16bf*)(sW + col * DFEAT + k0 + half * 16);
            acc[nt] = __builtin_amdgcn_wmma_f32_16x16x32_bf16(
                false, a, false, b, (short)0, acc[nt], false, false);
        }
    }

    // Store: acc vgpr v, lane L -> (row = rowbase + v + 8*half, col = nt*16 + lm)
#pragma unroll
    for (int nt = 0; nt < 8; ++nt) {
#pragma unroll
        for (int v = 0; v < 8; ++v) {
            int row = rowbase + v + half * 8;
            float x = acc[nt][v];
            if (RELU) x = fmaxf(x, 0.0f);
            if (row < nrows) out[(size_t)row * DFEAT + nt * 16 + lm] = x;
        }
    }
}

// ---- mean pool helpers ----
__global__ void k_zero128(float* p) {
    if (threadIdx.x < DFEAT) p[threadIdx.x] = 0.0f;
}

#define POOL_ROWS 250
__global__ void k_pool(const float* __restrict__ h, float* __restrict__ pooled, int nrows) {
    int c  = threadIdx.x;                       // 128 threads = one column each
    int r0 = blockIdx.x * POOL_ROWS;
    int r1 = r0 + POOL_ROWS; if (r1 > nrows) r1 = nrows;
    float s = 0.0f;
    for (int r = r0; r < r1; ++r) s += h[(size_t)r * DFEAT + c];
    atomic_add_f32_dev(pooled + c, s);
}

// ---- final fc: out[o] = (pooled/N) @ Wfc + bfc, O=64 ----
__global__ void k_fc(const float* __restrict__ pooled, const float* __restrict__ Wfc,
                     const float* __restrict__ bfc, float* __restrict__ out,
                     int nrows, int ocols) {
    int o = threadIdx.x;
    if (o >= ocols) return;
    float inv = 1.0f / (float)nrows;
    float s = bfc[o];
    for (int k = 0; k < DFEAT; ++k) s += pooled[k] * inv * Wfc[k * ocols + o];
    out[o] = s;
}

extern "C" void kernel_launch(void* const* d_in, const int* in_sizes, int n_in,
                              void* d_out, int out_size, void* d_ws, size_t ws_size,
                              hipStream_t stream) {
    const float*     x   = (const float*)d_in[0];
    const long long* ei  = (const long long*)d_in[1];
    const float*     W1a = (const float*)d_in[2];
    const float*     b1a = (const float*)d_in[3];
    const float*     W1b = (const float*)d_in[4];
    const float*     b1b = (const float*)d_in[5];
    const float*     W2a = (const float*)d_in[6];
    const float*     b2a = (const float*)d_in[7];
    const float*     W2b = (const float*)d_in[8];
    const float*     b2b = (const float*)d_in[9];
    const float*     Wfc = (const float*)d_in[10];
    const float*     bfc = (const float*)d_in[11];
    float*           out = (float*)d_out;

    const int N = in_sizes[0] / DFEAT;          // 100000
    const int E = in_sizes[1] / 2;              // 1600000
    const int O = out_size;                     // 64
    const long long src_off = 0, dst_off = E;   // edge_index rows

    // workspace layout
    char*   ws    = (char*)d_ws;
    size_t  fbyt  = (size_t)N * DFEAT * sizeof(float);
    float*  bufA  = (float*)(ws);
    float*  bufB  = (float*)(ws + fbyt);
    __bf16* wt    = (__bf16*)(ws + 2 * fbyt);   // 4 x 128x128 bf16 = 128 KB
    float*  pool  = (float*)(ws + 2 * fbyt + 4 * DFEAT * DFEAT * sizeof(__bf16));

    const int n4        = N * (DFEAT / 4);
    const int copyGrid  = (n4 + 255) / 256;
    const int scatGrid  = (E * 32 + 255) / 256;
    const int gemmGrid  = ((N + 15) / 16 + 7) / 8;      // 8 wave-tiles per block
    const int poolGrid  = (N + POOL_ROWS - 1) / POOL_ROWS;
    const int prepGrid  = (DFEAT * DFEAT + 127) / 128;

    // weights -> bf16, transposed
    hipLaunchKernelGGL(k_prep_wt, dim3(prepGrid), dim3(128), 0, stream, W1a, wt + 0 * DFEAT * DFEAT);
    hipLaunchKernelGGL(k_prep_wt, dim3(prepGrid), dim3(128), 0, stream, W1b, wt + 1 * DFEAT * DFEAT);
    hipLaunchKernelGGL(k_prep_wt, dim3(prepGrid), dim3(128), 0, stream, W2a, wt + 2 * DFEAT * DFEAT);
    hipLaunchKernelGGL(k_prep_wt, dim3(prepGrid), dim3(128), 0, stream, W2b, wt + 3 * DFEAT * DFEAT);

    // ---- layer 1 ----
    hipLaunchKernelGGL(k_copy,    dim3(copyGrid), dim3(256), 0, stream, (const float4*)x, (float4*)bufA, n4);
    hipLaunchKernelGGL(k_scatter, dim3(scatGrid), dim3(256), 0, stream, x, ei + src_off, ei + dst_off, bufA, E);
    hipLaunchKernelGGL((k_gemm<true>),  dim3(gemmGrid), dim3(256), 0, stream, bufA, wt + 0 * DFEAT * DFEAT, b1a, bufB, N);
    hipLaunchKernelGGL((k_gemm<false>), dim3(gemmGrid), dim3(256), 0, stream, bufB, wt + 1 * DFEAT * DFEAT, b1b, bufA, N);

    // ---- layer 2 ----  (h1 lives in bufA)
    hipLaunchKernelGGL(k_copy,    dim3(copyGrid), dim3(256), 0, stream, (const float4*)bufA, (float4*)bufB, n4);
    hipLaunchKernelGGL(k_scatter, dim3(scatGrid), dim3(256), 0, stream, bufA, ei + src_off, ei + dst_off, bufB, E);
    hipLaunchKernelGGL((k_gemm<true>),  dim3(gemmGrid), dim3(256), 0, stream, bufB, wt + 2 * DFEAT * DFEAT, b2a, bufA, N);
    hipLaunchKernelGGL((k_gemm<false>), dim3(gemmGrid), dim3(256), 0, stream, bufA, wt + 3 * DFEAT * DFEAT, b2b, bufB, N);

    // ---- mean pool + fc ----  (h2 lives in bufB)
    hipLaunchKernelGGL(k_zero128, dim3(1), dim3(128), 0, stream, pool);
    hipLaunchKernelGGL(k_pool,    dim3(poolGrid), dim3(128), 0, stream, bufB, pool, N);
    hipLaunchKernelGGL(k_fc,      dim3(1), dim3(64), 0, stream, pool, Wfc, bfc, out, N, O);
}
